// Denoiser_45535243272304
// MI455X (gfx1250) — matrix operand order repaired and checked by reading, not compiled, not run on hardware
//
#include <hip/hip_runtime.h>
#include <math.h>

// ---------------------------------------------------------------------------
// CDNA5 / gfx1250 S4-denoiser forward pass, round 2.
// wave32; WMMA f32_16x16x32_f16 for all contractions; weights pre-swizzled
// into ISA fragment order (contiguous 32B per lane); LDS planes transposed +
// padded so B-fragments are two aligned ds_load_b128; TDM async tensor load
// stages the xf tile; SSM kernel spectrum evaluated in closed form (damped
// cosine => two-pole geometric series), so no kernel FFT and no kf table.
// ---------------------------------------------------------------------------

typedef __attribute__((ext_vector_type(16))) _Float16 v16h;
typedef __attribute__((ext_vector_type(8)))  _Float16 v8h;
typedef __attribute__((ext_vector_type(8)))  float    v8f;
typedef __attribute__((ext_vector_type(4)))  unsigned int u32x4;
typedef __attribute__((ext_vector_type(8)))  int      i32x8;
typedef __attribute__((ext_vector_type(4)))  int      i32x4;

#define H0 ((_Float16)0.0f)
#define KP  264   // 256 + 8 halves pad: row stride 528B -> +16B bank rotation
#define AKP 520   // 512 + 8 halves pad: row stride 1040B

__device__ __forceinline__ v8f wmma32(v16h a, v16h b, v8f c) {
  return __builtin_amdgcn_wmma_f32_16x16x32_f16(false, a, false, b, (short)0, c,
                                                false, false);
}

union V16 { v16h v; v8h h[2]; };

// B fragment: 16 contiguous halves, 16B-aligned -> 2x ds_load_b128
__device__ __forceinline__ v16h ldB16(const _Float16* p) {
  V16 u;
  u.h[0] = *(const v8h*)(p);
  u.h[1] = *(const v8h*)(p + 8);
  return u.v;
}
// A fragment from an LDS row: chunks {k0+half8 .. +7} and {k0+16+half8 .. +7}
__device__ __forceinline__ v16h ldA16(const _Float16* p) {
  V16 u;
  u.h[0] = *(const v8h*)(p);
  u.h[1] = *(const v8h*)(p + 16);
  return u.v;
}

// ---- Tensor Data Mover: 2D f32 tile (tile_h rows x tile_w elems) -> LDS ----
__device__ __forceinline__ void tdm_load_2d_f32(unsigned lds_byte, const void* g,
                                                unsigned tile_w, unsigned tile_h,
                                                unsigned tensor_w,
                                                unsigned long long row_stride) {
  unsigned long long ga = (unsigned long long)g;
  u32x4 g0; i32x8 g1; i32x4 g2; i32x4 g3;
  g0[0] = 1u;
  g0[1] = lds_byte;
  g0[2] = (unsigned)(ga & 0xffffffffull);
  g0[3] = (unsigned)((ga >> 32) & 0x1ffffffull) | (2u << 30);
  g1[0] = (int)(2u << 16);                       // data_size = 4B
  g1[1] = (int)((tensor_w & 0xffffu) << 16);
  g1[2] = (int)(((tensor_w >> 16) & 0xffffu) | ((tile_h & 0xffffu) << 16));
  g1[3] = (int)(((tile_h >> 16) & 0xffffu) | ((tile_w & 0xffffu) << 16));
  g1[4] = (int)(tile_h & 0xffffu);
  g1[5] = (int)(row_stride & 0xffffffffull);
  g1[6] = (int)((row_stride >> 32) & 0xffffull);
  g1[7] = 0;
  g2[0] = g2[1] = g2[2] = g2[3] = 0;
  g3[0] = g3[1] = g3[2] = g3[3] = 0;
#if defined(__clang_major__) && __clang_major__ >= 23
  i32x8 g4 = {0, 0, 0, 0, 0, 0, 0, 0};
  __builtin_amdgcn_tensor_load_to_lds(g0, g1, g2, g3, g4, 0);
#else
  __builtin_amdgcn_tensor_load_to_lds(g0, g1, g2, g3, 0);
#endif
}

// ---- closed-form spectrum of K[l] = exp(-r l) cos(w l), l in [0,L) --------
__device__ __forceinline__ void kf_eval(float rho, float rhoL, float w, float th,
                                        float Lf, float* kr, float* ki) {
  float out_r = 0.f, out_i = 0.f;
#pragma unroll
  for (int s = 0; s < 2; ++s) {
    float phi = (s ? -w : w) - th;
    float sN, cN, s1, c1;
    __sincosf(phi * Lf, &sN, &cN);
    __sincosf(phi, &s1, &c1);
    float nr = 1.f - rhoL * cN, ni = -rhoL * sN;
    float dr = 1.f - rho * c1,  di = -rho * s1;
    float den = dr * dr + di * di + 1e-20f;
    out_r += (nr * dr + ni * di) / den;
    out_i += (ni * dr - nr * di) / den;
  }
  *kr = 0.5f * out_r;
  *ki = 0.5f * out_i;
}

// ===========================================================================
// Weight pre-swizzle into WMMA fragment order.
// A order: dst[(((mt*Ks+ks)*32+lane)*16)+j] = src[row,k] (*exp(logdt[row])),
//          row = mt*16+(lane&15), k = ks*32 + j + (lane&16)
// ===========================================================================
__global__ void k_pack_a(const float* __restrict__ src,
                         const float* __restrict__ logdt,
                         _Float16* __restrict__ dst, int M, int K, int Mt, int Ks) {
  long long i = (long long)blockIdx.x * blockDim.x + threadIdx.x;
  long long tot = (long long)Mt * Ks * 512;
  if (i >= tot) return;
  int j = (int)(i & 15), lane = (int)((i >> 4) & 31);
  long long tile = i >> 9;
  int ks = (int)(tile % Ks), mt = (int)(tile / Ks);
  int row = mt * 16 + (lane & 15);
  int k = ks * 32 + j + (lane & 16);
  float v = 0.f;
  if (row < M && k < K) {
    v = src[(size_t)row * K + k];
    if (logdt) v *= __expf(logdt[row]);
  }
  dst[i] = (_Float16)v;
}

// B order: col n = nt*16+(lane&15), k = ks*32 + ((j&8)<<1)+(j&7) + ((lane&16)>>1)
__global__ void k_pack_b_down(const float* __restrict__ W,
                              _Float16* __restrict__ dst,
                              int cin, int dch, int rf, int Nt, int Ks) {
  long long i = (long long)blockIdx.x * blockDim.x + threadIdx.x;
  long long tot = (long long)Nt * Ks * 512;
  if (i >= tot) return;
  int j = (int)(i & 15), lane = (int)((i >> 4) & 31);
  long long tile = i >> 9;
  int ks = (int)(tile % Ks), nt = (int)(tile / Ks);
  int n = nt * 16 + (lane & 15);
  int k = ks * 32 + ((j & 8) << 1) + (j & 7) + ((lane & 16) >> 1);
  float v = 0.f;
  if (n < dch && k < cin * rf) {
    int cc = k / rf, rr = k - cc * rf;
    v = W[((size_t)cc * dch + n) * rf + rr];
  }
  dst[i] = (_Float16)v;
}

__global__ void k_pack_b_up(const float* __restrict__ W,
                            _Float16* __restrict__ dst,
                            int cin, int dch, int rf, int Nt, int Ks) {
  long long i = (long long)blockIdx.x * blockDim.x + threadIdx.x;
  long long tot = (long long)Nt * Ks * 512;
  if (i >= tot) return;
  int j = (int)(i & 15), lane = (int)((i >> 4) & 31);
  long long tile = i >> 9;
  int ks = (int)(tile % Ks), nt = (int)(tile / Ks);
  int n = nt * 16 + (lane & 15);      // flattened (d, r)
  int k = ks * 32 + ((j & 8) << 1) + (j & 7) + ((lane & 16) >> 1);  // k = c
  float v = 0.f;
  if (n < dch * rf && k < cin) {
    int dd = n / rf, rr = n - dd * rf;
    v = W[((size_t)k * dch + dd) * rf + rr];
  }
  dst[i] = (_Float16)v;
}

// ===========================================================================
// Core SSM frequency-domain kernel.
//   u = Bhat(Nx c)*xf(cx16 cplx); v = kf .* u; yf = C(cxN)*v    [4 real GEMMs]
// LDS planes stored transposed: plane[reim][col(16)][k(KP)] f16.
// ===========================================================================
__global__ __launch_bounds__(256)
void k_freq_ssm(const float2* __restrict__ xf,
                const _Float16* __restrict__ pBhat,
                const _Float16* __restrict__ pC,
                const float* __restrict__ Ap, const float* __restrict__ logdt,
                float2* __restrict__ yf, int c, int n2, int Lker,
                int Ks1, int Mt2) {
  __shared__ __align__(128) char pool[17024 + 32768];
  _Float16* xsm = (_Float16*)pool;                 // [2][16][KP]
  float2*  stage = (float2*)(pool + 17024);        // [256*16] f32 staging
  _Float16* vsm = (_Float16*)(pool + 17024);       // [2][16][KP] (alias stage)

  const int ftile = blockIdx.x, b = blockIdx.y;
  const int tid = threadIdx.x, lane = tid & 31, wave = tid >> 5;
  const int f0 = ftile * 16;
  const int half8 = (lane & 16) >> 1;
  const int bK16  = (lane & 16);
  const int row15 = lane & 15;

  // ---- Phase 0: TDM DMA of the xf tile (c rows x 16 complex) --------------
  if (wave == 0) {
    const float2* src = xf + (size_t)b * c * n2 + f0;
    tdm_load_2d_f32((unsigned)(unsigned long long)(void*)stage, src,
                    32u, (unsigned)c, (unsigned)(2 * n2),
                    (unsigned long long)(2 * n2));
    __builtin_amdgcn_s_wait_tensorcnt(0);
  }
  __syncthreads();

  // transpose-convert into xsm[reim][col][k], zero-pad k >= c (select, no branch)
  for (int i = tid; i < 256 * 16; i += 256) {
    int row = i >> 4, col = i & 15;
    float2 v = stage[row * 16 + col];
    bool ok = row < c;
    xsm[(0 * 16 + col) * KP + row] = (_Float16)(ok ? v.x : 0.f);
    xsm[(1 * 16 + col) * KP + row] = (_Float16)(ok ? v.y : 0.f);
  }
  __syncthreads();

  // ---- Phase 1: u = Bhat * xf; v = kf .* u -> LDS -------------------------
  for (int mt = wave; mt < 16; mt += 8) {
    v8f ar = {}, ai = {};
    const _Float16* wa = pBhat + ((size_t)(mt * Ks1) * 32 + lane) * 16;
    const _Float16* xr = &xsm[(0 * 16 + row15) * KP + bK16];
    const _Float16* xi = &xsm[(1 * 16 + row15) * KP + bK16];
    for (int ks = 0; ks < Ks1; ++ks) {
      v16h a  = *(const v16h*)(wa + (size_t)ks * 512);
      v16h br = ldB16(xr + ks * 32);
      v16h bi = ldB16(xi + ks * 32);
      ar = wmma32(a, br, ar);
      ai = wmma32(a, bi, ai);
    }
#pragma unroll
    for (int i = 0; i < 8; ++i) {
      int M  = i + ((lane >> 4) << 3);
      int nn = mt * 16 + M;
      int fc = lane & 15;
      float th  = 6.28318530718f * (float)(f0 + fc) / (float)n2;
      float dt2 = __expf(logdt[nn]);
      float r   = dt2 * log1pf(__expf(Ap[nn * 2 + 0]));
      float w   = dt2 * Ap[nn * 2 + 1];
      float kr, ki;
      kf_eval(__expf(-r), __expf(-r * (float)Lker), w, th, (float)Lker, &kr, &ki);
      float ur = ar[i], ui = ai[i];
      vsm[(0 * 16 + fc) * KP + nn] = (_Float16)(kr * ur - ki * ui);
      vsm[(1 * 16 + fc) * KP + nn] = (_Float16)(kr * ui + ki * ur);
    }
  }
  __syncthreads();

  // ---- Phase 2: yf = C * v ------------------------------------------------
  for (int dt_ = wave; dt_ < Mt2; dt_ += 8) {
    v8f yr = {}, yi = {};
    const _Float16* wc = pC + ((size_t)(dt_ * 8) * 32 + lane) * 16;
    const _Float16* vr = &vsm[(0 * 16 + row15) * KP + bK16];
    const _Float16* vi = &vsm[(1 * 16 + row15) * KP + bK16];
#pragma unroll
    for (int ks = 0; ks < 8; ++ks) {
      v16h a  = *(const v16h*)(wc + (size_t)ks * 512);
      v16h br = ldB16(vr + ks * 32);
      v16h bi = ldB16(vi + ks * 32);
      yr = wmma32(a, br, yr);
      yi = wmma32(a, bi, yi);
    }
#pragma unroll
    for (int i = 0; i < 8; ++i) {
      int dd = dt_ * 16 + i + ((lane >> 4) << 3);
      int fc = lane & 15;
      if (dd < c)
        yf[((size_t)b * c + dd) * n2 + f0 + fc] = make_float2(yr[i], yi[i]);
    }
  }
}

// ===========================================================================
// Channel mix (down): y[b,d,t] = sum_{c,r} x[b,c,t*rf+r] W[c,d,r]
// One wave per 16x16 tile; A tile staged in LDS (zero-filled, clamp-indexed);
// W pre-swizzled into B-fragment order.
// ===========================================================================
__global__ __launch_bounds__(32)
void k_mix_down(const float* __restrict__ X, const _Float16* __restrict__ Wp,
                float* __restrict__ Y, int cin, int Lin, int dch, int rf,
                int Ks) {
  __shared__ __align__(16) _Float16 Al[16 * AKP];
  const int T = Lin / rf, K = cin * rf, Kp = Ks * 32;
  const int tt = blockIdx.x, dti = blockIdx.y, b = blockIdx.z;
  const int lane = threadIdx.x;
  const int half8 = (lane & 16) >> 1, row15 = lane & 15;

  v8h z8 = {H0, H0, H0, H0, H0, H0, H0, H0};
  for (int i = lane; i < 16 * (Kp >> 3); i += 32) {
    int t = i / (Kp >> 3), q = i - t * (Kp >> 3);
    *(v8h*)&Al[t * AKP + q * 8] = z8;
  }
  for (int i = lane; i < 16 * K; i += 32) {
    int t = i / K, k = i - t * K;
    int tg = tt * 16 + t;
    int cc = k / rf, rr = k - cc * rf;
    int tc = tg < T ? tg : T - 1;                 // clamp: load unconditional
    float v = X[((size_t)b * cin + cc) * Lin + (size_t)tc * rf + rr];
    Al[t * AKP + k] = (_Float16)(tg < T ? v : 0.f);
  }
  __syncthreads();

  v8f acc = {};
  const _Float16* wb = Wp + ((size_t)(dti * Ks) * 32 + lane) * 16;
  const _Float16* arow = &Al[row15 * AKP + half8];
  for (int ks = 0; ks < Ks; ++ks) {
    v16h bf = *(const v16h*)(wb + (size_t)ks * 512);
    v16h a  = ldA16(arow + ks * 32);
    acc = wmma32(a, bf, acc);
  }
#pragma unroll
  for (int i = 0; i < 8; ++i) {
    int t = tt * 16 + i + ((lane >> 4) << 3);
    int d = dti * 16 + (lane & 15);
    if (t < T && d < dch) Y[((size_t)b * dch + d) * T + t] = acc[i];
  }
}

// ===========================================================================
// Channel mix (up): y[b,d,t*rf+r] = sum_c x[b,c,t] W[c,d,r]; cols j = d*rf+r
// ===========================================================================
__global__ __launch_bounds__(32)
void k_mix_up(const float* __restrict__ X, const _Float16* __restrict__ Wp,
              float* __restrict__ Y, int cin, int T, int dch, int rf, int Ks) {
  __shared__ __align__(16) _Float16 Al[16 * KP];
  const int NC = dch * rf, Kp = Ks * 32;
  const int tt = blockIdx.x, jt = blockIdx.y, b = blockIdx.z;
  const int lane = threadIdx.x;
  const int half8 = (lane & 16) >> 1, row15 = lane & 15;

  v8h z8 = {H0, H0, H0, H0, H0, H0, H0, H0};
  for (int i = lane; i < 16 * (Kp >> 3); i += 32) {
    int t = i / (Kp >> 3), q = i - t * (Kp >> 3);
    *(v8h*)&Al[t * KP + q * 8] = z8;
  }
  for (int i = lane; i < 16 * cin; i += 32) {
    int t = i / cin, k = i - t * cin;
    int tg = tt * 16 + t;
    int tc = tg < T ? tg : T - 1;
    float v = X[((size_t)b * cin + k) * T + tc];
    Al[t * KP + k] = (_Float16)(tg < T ? v : 0.f);
  }
  __syncthreads();

  v8f acc = {};
  const _Float16* wb = Wp + ((size_t)(jt * Ks) * 32 + lane) * 16;
  const _Float16* arow = &Al[row15 * KP + half8];
  for (int ks = 0; ks < Ks; ++ks) {
    v16h bf = *(const v16h*)(wb + (size_t)ks * 512);
    v16h a  = ldA16(arow + ks * 32);
    acc = wmma32(a, bf, acc);
  }
#pragma unroll
  for (int i = 0; i < 8; ++i) {
    int t = tt * 16 + i + ((lane >> 4) << 3);
    int jc = jt * 16 + (lane & 15);
    if (t < T && jc < NC) {
      int dd = jc / rf, rr = jc - dd * rf;
      Y[((size_t)b * dch + dd) * ((size_t)T * rf) + (size_t)t * rf + rr] = acc[i];
    }
  }
}

// ===========================================================================
// In-LDS radix-2 complex FFT (one strided sub-transform per block).
// ===========================================================================
__global__ void k_fft(float2* __restrict__ d, int n, int logn, int stride,
                      int sub_stride, long long seqlen, int dir, float scale) {
  extern __shared__ float2 fsm[];
  float2* base = d + (size_t)blockIdx.y * seqlen + (size_t)blockIdx.x * sub_stride;
  for (int i = threadIdx.x; i < n; i += blockDim.x) {
    int j = (int)(__brev((unsigned)i) >> (32 - logn));
    fsm[j] = base[(size_t)i * stride];
  }
  __syncthreads();
  for (int s = 1; s <= logn; ++s) {
    int m = 1 << s, hm = m >> 1;
    float ang0 = (dir > 0 ? -6.28318530718f : 6.28318530718f) / (float)m;
    for (int idx = threadIdx.x; idx < (n >> 1); idx += blockDim.x) {
      int grp = idx / hm, pos = idx - grp * hm;
      int i0 = grp * m + pos, i1 = i0 + hm;
      float sa, ca;
      __sincosf(ang0 * (float)pos, &sa, &ca);
      float2 u = fsm[i0], v = fsm[i1];
      float tr = ca * v.x - sa * v.y, ti = ca * v.y + sa * v.x;
      fsm[i0] = make_float2(u.x + tr, u.y + ti);
      fsm[i1] = make_float2(u.x - tr, u.y - ti);
    }
    __syncthreads();
  }
  for (int i = threadIdx.x; i < n; i += blockDim.x) {
    float2 v = fsm[i];
    base[(size_t)i * stride] = make_float2(v.x * scale, v.y * scale);
  }
}

__global__ void k_twiddle(float2* __restrict__ d, int n1, int n2c,
                          long long seqlen, int dir) {
  long long i = (long long)blockIdx.x * blockDim.x + threadIdx.x;
  long long n = (long long)n1 * n2c;
  if (i >= n) return;
  int i1 = (int)(i / n2c), i2 = (int)(i % n2c);
  float ang = (dir > 0 ? -6.28318530718f : 6.28318530718f) *
              ((float)((long long)i1 * i2) / (float)n);
  float sa, ca;
  __sincosf(ang, &sa, &ca);
  size_t idx = (size_t)blockIdx.y * seqlen + i;
  float2 v = d[idx];
  d[idx] = make_float2(v.x * ca - v.y * sa, v.x * sa + v.y * ca);
}

// ---------------- elementwise helpers --------------------------------------
__global__ void k_pack(const float* __restrict__ x, float2* __restrict__ X,
                       long long nseq, int L, int n2) {
  long long i = (long long)blockIdx.x * blockDim.x + threadIdx.x;
  if (i >= nseq * (long long)n2) return;
  long long s = i / n2;
  int t = (int)(i % n2);
  float v = (t < L) ? x[s * L + t] : 0.f;
  X[i] = make_float2(v, 0.f);
}

__global__ void k_extract(const float2* __restrict__ Y, float* __restrict__ y,
                          long long nseq, int L, int n2) {
  long long i = (long long)blockIdx.x * blockDim.x + threadIdx.x;
  if (i >= nseq * (long long)L) return;
  long long s = i / L;
  int t = (int)(i % L);
  y[i] = Y[s * (long long)n2 + t].x;
}

__global__ void k_add(float* __restrict__ y, const float* __restrict__ s,
                      long long n) {
  long long i = (long long)blockIdx.x * blockDim.x + threadIdx.x;
  if (i < n) y[i] += s[i];
}

__global__ void k_dwconv3(const float* __restrict__ X, const float* __restrict__ w,
                          const float* __restrict__ bias, float* __restrict__ Y,
                          int Bn, int c, int L) {
  long long i = (long long)blockIdx.x * blockDim.x + threadIdx.x;
  if (i >= (long long)Bn * c * L) return;
  int t = (int)(i % L);
  int cc = (int)((i / L) % c);
  float xm = (t > 0) ? X[i - 1] : 0.f;
  float x0 = X[i];
  float xp = (t < L - 1) ? X[i + 1] : 0.f;
  Y[i] = w[cc * 3 + 0] * xm + w[cc * 3 + 1] * x0 + w[cc * 3 + 2] * xp + bias[cc];
}

__global__ void k_ln_silu(float* __restrict__ X, const float* __restrict__ g,
                          const float* __restrict__ bt, int Bn, int c, int L,
                          int use_ln) {
  long long idx = (long long)blockIdx.x * blockDim.x + threadIdx.x;
  if (idx >= (long long)Bn * L) return;
  int b = (int)(idx / L), t = (int)(idx % L);
  float* col = X + (size_t)b * c * L + t;
  if (use_ln) {
    float m = 0.f, v = 0.f;
    for (int i = 0; i < c; ++i) m += col[(size_t)i * L];
    m /= (float)c;
    for (int i = 0; i < c; ++i) { float d = col[(size_t)i * L] - m; v += d * d; }
    float inv = rsqrtf(v / (float)c + 1e-5f);
    for (int i = 0; i < c; ++i) {
      float y = (col[(size_t)i * L] - m) * inv * g[i] + bt[i];
      col[(size_t)i * L] = y / (1.f + __expf(-y));
    }
  } else {
    for (int i = 0; i < c; ++i) {
      float y = col[(size_t)i * L];
      col[(size_t)i * L] = y / (1.f + __expf(-y));
    }
  }
}

// ===========================================================================
// Host orchestration
// ===========================================================================
struct BlockP {
  const float *cw, *cb, *A, *B, *ld, *C, *g, *bt;
  _Float16 *pB, *pC;
  int ch, Ks1, Mt2;
};

static inline int ilog2i(int n) { int l = 0; while ((1 << l) < n) ++l; return l; }

static void fft_any(float2* d, int nseq, int n, int dir, float scale,
                    hipStream_t s) {
  if (n <= 4096) {
    k_fft<<<dim3(1, nseq), 256, (size_t)n * sizeof(float2), s>>>(
        d, n, ilog2i(n), 1, 0, (long long)n, dir, scale);
  } else {  // four-step: n = n1 x 4096 (bins permuted; GEMM stage is per-bin)
    int n2c = 4096, n1 = n / n2c;
    k_fft<<<dim3(n2c, nseq), 256, (size_t)n1 * sizeof(float2), s>>>(
        d, n1, ilog2i(n1), n2c, 1, (long long)n, dir, 1.0f);
    k_twiddle<<<dim3((unsigned)((n + 255) / 256), nseq), 256, 0, s>>>(
        d, n1, n2c, (long long)n, dir);
    k_fft<<<dim3(n1, nseq), 256, (size_t)n2c * sizeof(float2), s>>>(
        d, n2c, ilog2i(n2c), 1, n2c, (long long)n, dir, scale);
  }
}

static void apply_block(const BlockP& bp, float* x, float* tmp, float2* xc,
                        float2* yc, int Bn, int L, int act, hipStream_t s) {
  int c = bp.ch, n2 = 2 * L;
  long long nseq = (long long)Bn * c;
  const float* sin_ = x;
  if (bp.cw) {
    long long tot = nseq * L;
    k_dwconv3<<<(unsigned)((tot + 255) / 256), 256, 0, s>>>(x, bp.cw, bp.cb, tmp,
                                                            Bn, c, L);
    sin_ = tmp;
  }
  long long tp = nseq * n2;
  k_pack<<<(unsigned)((tp + 255) / 256), 256, 0, s>>>(sin_, xc, nseq, L, n2);
  fft_any(xc, (int)nseq, n2, +1, 1.0f, s);
  k_freq_ssm<<<dim3(n2 / 16, Bn), 256, 0, s>>>(xc, bp.pB, bp.pC, bp.A, bp.ld,
                                               yc, c, n2, L, bp.Ks1, bp.Mt2);
  fft_any(yc, (int)nseq, n2, -1, 1.0f / (float)n2, s);
  long long te = nseq * L;
  k_extract<<<(unsigned)((te + 255) / 256), 256, 0, s>>>(yc, x, nseq, L, n2);
  if (act) {
    long long tl = (long long)Bn * L;
    k_ln_silu<<<(unsigned)((tl + 255) / 256), 256, 0, s>>>(x, bp.g, bp.bt, Bn, c,
                                                           L, bp.g ? 1 : 0);
  }
}

extern "C" void kernel_launch(void* const* d_in, const int* in_sizes, int n_in,
                              void* d_out, int out_size, void* d_ws,
                              size_t ws_size, hipStream_t stream) {
  (void)in_sizes; (void)n_in; (void)ws_size;
  const int CH[7] = {1, 16, 32, 64, 96, 128, 256};
  const int RF[6] = {4, 4, 2, 2, 2, 2};
  const long long Ls[6] = {65536, 16384, 4096, 2048, 1024, 512};
  const int Bn = 32, L0 = 65536;

  char* basep = (char*)d_ws;
  size_t off = 0;
  auto alloc = [&](size_t bytes) -> void* {
    off = (off + 255) & ~(size_t)255;
    void* p = basep + off;
    off += bytes;
    return p;
  };
  size_t maxact = (size_t)Bn * 16 * 16384;
  float* bufA = (float*)alloc(maxact * 4);
  float* bufB = (float*)alloc(maxact * 4);
  float* bufC = (float*)alloc(maxact * 4);
  size_t maxcpx = (size_t)Bn * 16 * 32768;
  float2* xc = (float2*)alloc(maxcpx * 8);
  float2* yc = (float2*)alloc(maxcpx * 8);
  float* skip[6];
  for (int i = 0; i < 6; ++i)
    skip[i] = (float*)alloc((size_t)Bn * CH[i] * Ls[i] * 4);

  // ---- walk the flattened params ------------------------------------------
  int pi = 1;
  auto nxt = [&]() -> const float* { return (const float*)d_in[pi++]; };
  auto readB = [&](int ch, bool act_) {
    BlockP b = {};
    b.ch = ch;
    if (ch > 1) { b.cw = nxt(); b.cb = nxt(); }
    b.A = nxt(); b.B = nxt(); b.ld = nxt(); b.C = nxt();
    if (act_ && ch > 1) { b.g = nxt(); b.bt = nxt(); }
    return b;
  };
  BlockP downB[6]; const float* downW[6];
  for (int i = 0; i < 6; ++i) { downB[i] = readB(CH[i], true); downW[i] = nxt(); }
  BlockP hidB[2];
  for (int h = 0; h < 2; ++h) hidB[h] = readB(256, true);
  const float* upW[6]; BlockP upB[6];
  for (int i = 0; i < 6; ++i) { upW[i] = nxt(); upB[i] = readB(CH[i], true); }
  BlockP lastB[2];
  lastB[0] = readB(1, true);
  lastB[1] = readB(1, false);

  // ---- weight pre-swizzle -------------------------------------------------
  auto prep_block = [&](BlockP& bp) {
    int cc = bp.ch;
    bp.Ks1 = (cc + 31) / 32;
    bp.Mt2 = (cc + 15) / 16;
    size_t nb = (size_t)16 * bp.Ks1 * 512;
    size_t nc = (size_t)bp.Mt2 * 8 * 512;
    bp.pB = (_Float16*)alloc(nb * 2);
    bp.pC = (_Float16*)alloc(nc * 2);
    k_pack_a<<<(unsigned)((nb + 255) / 256), 256, 0, stream>>>(
        bp.B, bp.ld, bp.pB, 256, cc, 16, bp.Ks1);
    k_pack_a<<<(unsigned)((nc + 255) / 256), 256, 0, stream>>>(
        bp.C, (const float*)nullptr, bp.pC, cc, 256, bp.Mt2, 8);
  };
  for (int i = 0; i < 6; ++i) prep_block(downB[i]);
  for (int h = 0; h < 2; ++h) prep_block(hidB[h]);
  for (int i = 0; i < 6; ++i) prep_block(upB[i]);
  prep_block(lastB[0]);
  prep_block(lastB[1]);

  _Float16* pWd[6]; int Ksd[6];
  _Float16* pWu[6]; int Ksu[6];
  for (int i = 0; i < 6; ++i) {
    int cin = CH[i], dch = CH[i + 1], rf = RF[i];
    Ksd[i] = (cin * rf + 31) / 32;
    int Nt = (dch + 15) / 16;
    size_t n = (size_t)Nt * Ksd[i] * 512;
    pWd[i] = (_Float16*)alloc(n * 2);
    k_pack_b_down<<<(unsigned)((n + 255) / 256), 256, 0, stream>>>(
        downW[i], pWd[i], cin, dch, rf, Nt, Ksd[i]);
  }
  for (int i = 0; i < 6; ++i) {
    int cin = CH[i + 1], dch = CH[i], rf = RF[i];
    Ksu[i] = (cin + 31) / 32;
    int Nt = (dch * rf + 15) / 16;
    size_t n = (size_t)Nt * Ksu[i] * 512;
    pWu[i] = (_Float16*)alloc(n * 2);
    k_pack_b_up<<<(unsigned)((n + 255) / 256), 256, 0, stream>>>(
        upW[i], pWu[i], cin, dch, rf, Nt, Ksu[i]);
  }

  // ---- forward pass -------------------------------------------------------
  float* cur = bufA; float* alt = bufB;
  int c = 1, L = L0;
  hipMemcpyAsync(cur, d_in[0], (size_t)Bn * L0 * sizeof(float),
                 hipMemcpyDeviceToDevice, stream);

  for (int i = 0; i < 6; ++i) {  // down path
    hipMemcpyAsync(skip[i], cur, (size_t)Bn * c * L * sizeof(float),
                   hipMemcpyDeviceToDevice, stream);
    apply_block(downB[i], cur, bufC, xc, yc, Bn, L, 1, stream);
    int dch = CH[i + 1], rf = RF[i], T = L / rf;
    dim3 g((unsigned)((T + 15) / 16), (unsigned)((dch + 15) / 16), (unsigned)Bn);
    k_mix_down<<<g, 32, 0, stream>>>(cur, pWd[i], alt, c, L, dch, rf, Ksd[i]);
    float* t_ = cur; cur = alt; alt = t_;
    c = dch; L = T;
  }
  for (int h = 0; h < 2; ++h)    // bottleneck
    apply_block(hidB[h], cur, bufC, xc, yc, Bn, L, 1, stream);
  for (int j = 5; j >= 0; --j) { // up path
    int cin = CH[j + 1], dch = CH[j], rf = RF[j];
    dim3 g((unsigned)((L + 15) / 16), (unsigned)((dch * rf + 15) / 16),
           (unsigned)Bn);
    k_mix_up<<<g, 32, 0, stream>>>(cur, pWu[j], alt, cin, L, dch, rf, Ksu[j]);
    float* t_ = cur; cur = alt; alt = t_;
    c = dch; L = L * rf;
    long long tot = (long long)Bn * c * L;
    k_add<<<(unsigned)((tot + 255) / 256), 256, 0, stream>>>(cur, skip[j], tot);
    apply_block(upB[j], cur, bufC, xc, yc, Bn, L, 1, stream);
  }
  apply_block(lastB[0], cur, bufC, xc, yc, Bn, L, 1, stream);
  apply_block(lastB[1], cur, bufC, xc, yc, Bn, L, 0, stream);

  hipMemcpyAsync(d_out, cur, (size_t)out_size * sizeof(float),
                 hipMemcpyDeviceToDevice, stream);
}
